// Qwen3VisionTransformer_60447369724197
// MI455X (gfx1250) — compile-verified, matrix operand order checked by reading, not run on hardware
//
#include <hip/hip_runtime.h>

typedef __bf16 bf16;
typedef __attribute__((ext_vector_type(16))) __bf16 v16bf;
typedef __attribute__((ext_vector_type(8))) float v8f;

#define SEQn   2304
#define HIDn   1152
#define HEADSn 16
#define HDn    72
#define QKVn   3456
#define INTn   4304
#define MROWSn 576
#define MHIDn  4608
#define OUTn   2048

// ---------------------------------------------------------------------------
// Window-shuffle permutation (closed form).
__device__ __forceinline__ int widx_fwd(int og) {
  int b = og & 3, a = (og >> 2) & 3, nwi = (og >> 4) % 6, nhi = og / 96;
  return (nhi * 4 + a) * 24 + (nwi * 4 + b);
}
__device__ __forceinline__ int widx_inv(int g) {
  int gh = g / 24, gw = g % 24;
  return (((gh >> 2) * 6 + (gw >> 2)) * 4 + (gh & 3)) * 4 + (gw & 3);
}

// ---------------------------------------------------------------------------
// WMMA helpers.  A-frag (16x32 bf16): lane<16 = row lane, K chunks
// [0..7],[16..23]; lane>=16 = K chunks [8..15],[24..31].  B-frag is the same
// layout on the N-major (transposed) operand.  Each chunk = one ds_load_b128.
__device__ __forceinline__ v16bf ldfrag(const bf16* p, int khalf) {
  union { v16bf v; uint4 u[2]; } t;
  t.u[0] = *(const uint4*)(p + khalf);
  t.u[1] = *(const uint4*)(p + 16 + khalf);
  return t.v;
}

__device__ __forceinline__ v8f wmma_bf16(v16bf a, v16bf b, v8f c) {
  return __builtin_amdgcn_wmma_f32_16x16x32_bf16(false, a, false, b, (short)0,
                                                 c, false, false);
}

// 4 bf16 packed into one 8-byte LDS store.
__device__ __forceinline__ void store4bf(bf16* p, float a, float b, float c,
                                         float d) {
  union { bf16 h[4]; uint2 u; } t;
  t.h[0] = (bf16)a; t.h[1] = (bf16)b; t.h[2] = (bf16)c; t.h[3] = (bf16)d;
  *(uint2*)p = t.u;
}

__device__ __forceinline__ float redmax16(float v) {
#pragma unroll
  for (int m = 1; m < 16; m <<= 1) v = fmaxf(v, __shfl_xor(v, m, 32));
  return v;
}
__device__ __forceinline__ float redsum16(float v) {
#pragma unroll
  for (int m = 1; m < 16; m <<= 1) v += __shfl_xor(v, m, 32);
  return v;
}

// ---------------------------------------------------------------------------
// bf16-WMMA GEMM: C[M,N] = epi(A[M,K] @ W[K,N] + bias, aux) with optional
// output-row permutation.  Block tile 128x128x64, 256 threads = 8 waves in a
// 4(M) x 2(N) grid; each wave owns 32x64 of C (2 M-frags x 4 N-frags) and
// issues 16 WMMAs per staging stage.  f32 -> bf16 conversion happens while
// staging into LDS; W is transposed via register 4x4 blocks so every LDS
// store is a packed b64.  All K values are multiples of 4, so tiles are
// whole-float4 guarded.
enum { EPI_NONE, EPI_RES, EPI_SILU, EPI_MUL, EPI_GELU };
enum { PERM_NONE, PERM_G4, PERM_MERGED };

template <int EPI, int PERM>
__global__ __launch_bounds__(256) void gemm_bf16(
    const float* __restrict__ A, const float* __restrict__ W,
    const float* __restrict__ bias, const float* __restrict__ aux,
    float* __restrict__ C, int M, int N, int K) {
  __shared__ __align__(16) bf16 As[128][72];  // 64 + 8 pad, 16B-aligned rows
  __shared__ __align__(16) bf16 Ws[128][72];  // N-major (transposed)
  const int bm = blockIdx.y * 128;
  const int bn = blockIdx.x * 128;
  const int tid = threadIdx.x;
  const int lane = tid & 31, wid = tid >> 5;
  const int wm = (wid & 3) * 32, wn = (wid >> 2) * 64;
  const int rl = lane & 15, khalf = (lane >> 4) * 8;

  v8f acc[2][4];
#pragma unroll
  for (int mi = 0; mi < 2; ++mi)
#pragma unroll
    for (int nt = 0; nt < 4; ++nt)
#pragma unroll
      for (int e = 0; e < 8; ++e) acc[mi][nt][e] = 0.0f;

  for (int k0 = 0; k0 < K; k0 += 64) {
    __syncthreads();
    // stage A 128x64 (f32 -> bf16): 8 float4 per thread, packed b64 stores
#pragma unroll
    for (int it = 0; it < 8; ++it) {
      int idx = tid + it * 256;
      int r = idx >> 4, c4 = (idx & 15) << 2;  // 16 float4 per 64-wide row
      int gk = k0 + c4;
      float x0 = 0.f, x1 = 0.f, x2 = 0.f, x3 = 0.f;
      if ((bm + r) < M && gk + 4 <= K) {
        float4 v = *(const float4*)(A + (size_t)(bm + r) * K + gk);
        x0 = v.x; x1 = v.y; x2 = v.z; x3 = v.w;
      }
      store4bf(&As[r][c4], x0, x1, x2, x3);
    }
    // stage W 64x128 transposed via register 4x4 blocks: 2 blocks per thread
#pragma unroll
    for (int it = 0; it < 2; ++it) {
      int bidx = tid + it * 256;          // 512 blocks of 4x4
      int kb = bidx >> 5, nb = bidx & 31; // 16 k-blocks x 32 n-blocks
      int gk0 = k0 + kb * 4, gn0 = bn + nb * 4;
      float4 row[4];
#pragma unroll
      for (int i = 0; i < 4; ++i) {
        row[i] = make_float4(0.f, 0.f, 0.f, 0.f);
        if (gk0 + i < K && gn0 < N)  // N % 4 == 0 -> whole float4 in bounds
          row[i] = *(const float4*)(W + (size_t)(gk0 + i) * N + gn0);
      }
      int nloc = nb * 4;
      store4bf(&Ws[nloc + 0][kb * 4], row[0].x, row[1].x, row[2].x, row[3].x);
      store4bf(&Ws[nloc + 1][kb * 4], row[0].y, row[1].y, row[2].y, row[3].y);
      store4bf(&Ws[nloc + 2][kb * 4], row[0].z, row[1].z, row[2].z, row[3].z);
      store4bf(&Ws[nloc + 3][kb * 4], row[0].w, row[1].w, row[2].w, row[3].w);
    }
    __syncthreads();

#pragma unroll
    for (int ks = 0; ks < 2; ++ks) {
      v16bf a0 = ldfrag(&As[wm + rl][ks * 32], khalf);
      v16bf a1 = ldfrag(&As[wm + 16 + rl][ks * 32], khalf);
#pragma unroll
      for (int nt = 0; nt < 4; ++nt) {
        v16bf bfr = ldfrag(&Ws[wn + nt * 16 + rl][ks * 32], khalf);
        acc[0][nt] = wmma_bf16(a0, bfr, acc[0][nt]);
        acc[1][nt] = wmma_bf16(a1, bfr, acc[1][nt]);
      }
    }
  }

  // epilogue: C layout => VGPR e is rows (base+e) and (base+e+8), lanes = cols
#pragma unroll
  for (int mi = 0; mi < 2; ++mi) {
    const int rbase = bm + wm + mi * 16 + ((lane >> 4) ? 8 : 0);
#pragma unroll
    for (int nt = 0; nt < 4; ++nt) {
      int col = bn + wn + nt * 16 + rl;
      if (col >= N) continue;
      float bv = bias ? bias[col] : 0.0f;
#pragma unroll
      for (int e = 0; e < 8; ++e) {
        int row = rbase + e;
        if (row >= M) continue;
        float v = acc[mi][nt][e] + bv;
        if (EPI == EPI_RES)  v += aux[(size_t)row * N + col];
        if (EPI == EPI_SILU) v = v / (1.0f + __expf(-v));
        if (EPI == EPI_MUL)  v = v * aux[(size_t)row * N + col];
        if (EPI == EPI_GELU)
          v = 0.5f * v *
              (1.0f + tanhf(0.7978845608f * (v + 0.044715f * v * v * v)));
        int orow = row;
        if (PERM == PERM_G4)     orow = widx_inv(row >> 2) * 4 + (row & 3);
        if (PERM == PERM_MERGED) orow = widx_fwd(row);
        C[(size_t)orow * N + col] = v;
      }
    }
  }
}

// ---------------------------------------------------------------------------
// Flash-style fused attention: one block (4 waves) per (64-query tile, head).
// Streams 64-key KV blocks through LDS with online softmax; head_dim 72 is
// zero-padded to 96 (3 WMMA K-steps).  Per-row softmax stats are kept in the
// WMMA C-fragment lane layout so accumulator rescale needs no shuffles.
__global__ __launch_bounds__(128) void attn_kernel(
    const float* __restrict__ qkv, float* __restrict__ o, int win) {
  __shared__ __align__(16) bf16 Qs[64 * 104];  // 64 x 96 (+8 pad)
  __shared__ __align__(16) bf16 Ks[64 * 104];
  __shared__ __align__(16) bf16 Vt[96 * 72];   // d-major: Vt[d][k], 64 (+8)
  __shared__ __align__(16) bf16 Ps[64 * 72];   // P staging for A-frag reads
  const int qb = blockIdx.x, hh = blockIdx.y;
  const int tid = threadIdx.x, lane = tid & 31, wid = tid >> 5;
  const int q0 = qb * 64;
  const int kv0 = (q0 / win) * win;
  const int nkb = win / 64;
  const int wm = wid * 16;
  const int rl = lane & 15, khalf = (lane >> 4) * 8;
  const float scale = 0.1178511302f;  // 72^-0.5, folded into Q

  for (int idx = tid; idx < 64 * 96; idx += 128) {
    int r = idx / 96, d = idx % 96;
    float v =
        (d < HDn) ? qkv[(size_t)(q0 + r) * QKVn + hh * HDn + d] * scale : 0.0f;
    Qs[r * 104 + d] = (bf16)v;
  }

  v8f oacc[5];
#pragma unroll
  for (int t = 0; t < 5; ++t)
#pragma unroll
    for (int e = 0; e < 8; ++e) oacc[t][e] = 0.0f;
  float mrun[8], lrun[8];
#pragma unroll
  for (int e = 0; e < 8; ++e) { mrun[e] = -1e30f; lrun[e] = 0.0f; }

  for (int kb = 0; kb < nkb; ++kb) {
    const int k0 = kv0 + kb * 64;
    __syncthreads();  // previous iteration done with Ks/Vt (and Qs ready)
    for (int idx = tid; idx < 64 * 96; idx += 128) {
      int r = idx / 96, d = idx % 96;
      size_t base = (size_t)(k0 + r) * QKVn + hh * HDn + d;
      float kv_ = (d < HDn) ? qkv[base + HIDn] : 0.0f;
      Ks[r * 104 + d] = (bf16)kv_;
      float vv = (d < HDn) ? qkv[base + 2 * HIDn] : 0.0f;
      Vt[d * 72 + r] = (bf16)vv;
    }
    __syncthreads();

    // S = Q . K^T (contract over padded d = 96 -> 3 WMMA steps)
    v8f s[4];
#pragma unroll
    for (int nt = 0; nt < 4; ++nt)
#pragma unroll
      for (int e = 0; e < 8; ++e) s[nt][e] = 0.0f;
#pragma unroll
    for (int ks = 0; ks < 3; ++ks) {
      v16bf af = ldfrag(&Qs[(wm + rl) * 104 + ks * 32], khalf);
#pragma unroll
      for (int nt = 0; nt < 4; ++nt) {
        v16bf bfr = ldfrag(&Ks[(nt * 16 + rl) * 104 + ks * 32], khalf);
        s[nt] = wmma_bf16(af, bfr, s[nt]);
      }
    }

    // online softmax (row stats stay in C-fragment lane layout)
#pragma unroll
    for (int e = 0; e < 8; ++e) {
      float pm = fmaxf(fmaxf(s[0][e], s[1][e]), fmaxf(s[2][e], s[3][e]));
      pm = redmax16(pm);
      float mnew = fmaxf(mrun[e], pm);
      float sc = __expf(mrun[e] - mnew);
      mrun[e] = mnew;
      float ps = 0.0f;
#pragma unroll
      for (int nt = 0; nt < 4; ++nt) {
        float p = __expf(s[nt][e] - mnew);
        s[nt][e] = p;
        ps += p;
      }
      ps = redsum16(ps);
      lrun[e] = lrun[e] * sc + ps;
#pragma unroll
      for (int t = 0; t < 5; ++t) oacc[t][e] *= sc;
    }

    // stage P into LDS (wave-private rows; LDS is in-order per wave)
    const int qr = wm + ((lane >> 4) ? 8 : 0);
#pragma unroll
    for (int nt = 0; nt < 4; ++nt)
#pragma unroll
      for (int e = 0; e < 8; ++e)
        Ps[(qr + e) * 72 + nt * 16 + rl] = (bf16)s[nt][e];

    // O += P . V (2 WMMA K-steps over 64 keys, 5 d-tiles)
#pragma unroll
    for (int ks = 0; ks < 2; ++ks) {
      v16bf af = ldfrag(&Ps[(wm + rl) * 72 + ks * 32], khalf);
#pragma unroll
      for (int dt = 0; dt < 5; ++dt) {
        v16bf bfr = ldfrag(&Vt[(dt * 16 + rl) * 72 + ks * 32], khalf);
        oacc[dt] = wmma_bf16(af, bfr, oacc[dt]);
      }
    }
  }

  const int rbase = q0 + wm + ((lane >> 4) ? 8 : 0);
#pragma unroll
  for (int e = 0; e < 8; ++e) {
    float inv = 1.0f / lrun[e];
#pragma unroll
    for (int t = 0; t < 5; ++t) oacc[t][e] *= inv;
  }
#pragma unroll
  for (int dt = 0; dt < 5; ++dt) {
    int d = dt * 16 + rl;
    if (d < HDn) {
#pragma unroll
      for (int e = 0; e < 8; ++e)
        o[(size_t)(rbase + e) * HIDn + hh * HDn + d] = oacc[dt][e];
    }
  }
}

// ---------------------------------------------------------------------------
__global__ __launch_bounds__(256) void rmsnorm_kernel(
    const float* __restrict__ x, const float* __restrict__ g,
    float* __restrict__ out, int W) {
  const int row = blockIdx.x, tid = threadIdx.x;
  const float* xr = x + (size_t)row * W;
  float ss = 0.0f;
  for (int i = tid; i < W; i += 256) { float v = xr[i]; ss += v * v; }
#pragma unroll
  for (int m = 1; m < 32; m <<= 1) ss += __shfl_xor(ss, m, 32);
  __shared__ float red[8];
  if ((tid & 31) == 0) red[tid >> 5] = ss;
  __syncthreads();
  if (tid < 32) {
    float v = (tid < 8) ? red[tid] : 0.0f;
#pragma unroll
    for (int m = 1; m < 8; m <<= 1) v += __shfl_xor(v, m, 32);
    if (tid == 0) red[0] = v;
  }
  __syncthreads();
  const float r = rsqrtf(red[0] / (float)W + 1e-6f);
  float* orow = out + (size_t)row * W;
  for (int i = tid; i < W; i += 256) orow[i] = g[i] * xr[i] * r;
}

// cos/sin tables per shuffled sequence position (72 each; first 36 from hpos,
// last 36 from wpos, per the reference's concat layout).
__global__ void rope_tables_kernel(float* __restrict__ cosT,
                                   float* __restrict__ sinT) {
  const int s = blockIdx.x;
  const int t = threadIdx.x;
  if (t >= 72) return;
  int og = s >> 2, j = s & 3;
  int g = widx_fwd(og);
  int gh = g / 24, gw = g % 24;
  int hp = gh * 2 + (j >> 1);
  int wp = gw * 2 + (j & 1);
  int f = (t < 36) ? t : (t - 36);
  int pos = (t < 36) ? hp : wp;
  float inv = __expf(-(float)f * (9.210340371976184f / 36.0f));
  float ang = (float)pos * inv;
  cosT[(size_t)s * 72 + t] = __cosf(ang);
  sinT[(size_t)s * 72 + t] = __sinf(ang);
}

// In-place RoPE on the q and k thirds of the qkv buffer.
__global__ __launch_bounds__(256) void rope_apply_kernel(
    float* __restrict__ qkv, const float* __restrict__ cosT,
    const float* __restrict__ sinT) {
  const int s = blockIdx.x;
  const float* c = cosT + (size_t)s * 72;
  const float* sn = sinT + (size_t)s * 72;
  for (int idx = threadIdx.x; idx < HEADSn * 36; idx += 256) {
    int hd = idx / 36, j = idx % 36;
    float c0 = c[j], c1 = c[j + 36], s0 = sn[j], s1 = sn[j + 36];
    float* q = qkv + (size_t)s * QKVn + hd * HDn;
    float q0 = q[j], q1 = q[j + 36];
    q[j] = q0 * c0 - q1 * s0;
    q[j + 36] = q1 * c1 + q0 * s1;
    float* k = q + HIDn;
    float k0 = k[j], k1 = k[j + 36];
    k[j] = k0 * c0 - k1 * s0;
    k[j + 36] = k1 * c1 + k0 * s1;
  }
}

// ---------------------------------------------------------------------------
extern "C" void kernel_launch(void* const* d_in, const int* in_sizes, int n_in,
                              void* d_out, int out_size, void* d_ws,
                              size_t ws_size, hipStream_t stream) {
  const float* pixel    = (const float*)d_in[0];
  const float* patch_w  = (const float*)d_in[1];
  const float* norm1_g  = (const float*)d_in[2];
  const float* qkv_w    = (const float*)d_in[3];
  const float* qkv_b    = (const float*)d_in[4];
  const float* proj_w   = (const float*)d_in[5];
  const float* proj_b   = (const float*)d_in[6];
  const float* norm2_g  = (const float*)d_in[7];
  const float* gate_w   = (const float*)d_in[8];
  const float* gate_b   = (const float*)d_in[9];
  const float* up_w     = (const float*)d_in[10];
  const float* up_b     = (const float*)d_in[11];
  const float* down_w   = (const float*)d_in[12];
  const float* down_b   = (const float*)d_in[13];
  const float* m_norm_g = (const float*)d_in[14];
  const float* m_l1_w   = (const float*)d_in[15];
  const float* m_l1_b   = (const float*)d_in[16];
  const float* m_l2_w   = (const float*)d_in[17];
  const float* m_l2_b   = (const float*)d_in[18];
  const float* ds_norm_g= (const float*)d_in[19];
  const float* ds_l1_w  = (const float*)d_in[20];
  const float* ds_l1_b  = (const float*)d_in[21];
  const float* ds_l2_w  = (const float*)d_in[22];
  const float* ds_l2_b  = (const float*)d_in[23];
  float* out = (float*)d_out;

  // workspace (~105 MB of f32)
  float* ws   = (float*)d_ws;
  float* x    = ws;  ws += (size_t)SEQn * HIDn;   // residual stream
  float* hbuf = ws;  ws += (size_t)SEQn * HIDn;   // normed acts / merger m1
  float* qkvb = ws;  ws += (size_t)SEQn * QKVn;
  float* obuf = ws;  ws += (size_t)SEQn * HIDn;   // attn out / merger m2
  float* gbuf = ws;  ws += (size_t)SEQn * INTn;   // silu(gate), then gate*up
  float* cosT = ws;  ws += (size_t)SEQn * 72;
  float* sinT = ws;  ws += (size_t)SEQn * 72;

#define GRID(M_, N_) dim3(((N_) + 127) / 128, ((M_) + 127) / 128)

  rope_tables_kernel<<<SEQn, 96, 0, stream>>>(cosT, sinT);

  // patch embed, scattered into window-shuffled row order
  gemm_bf16<EPI_NONE, PERM_G4><<<GRID(SEQn, HIDn), 256, 0, stream>>>(
      pixel, patch_w, nullptr, nullptr, x, SEQn, HIDn, 1176);

  int ds_slot = 0;
  for (int i = 0; i < 8; ++i) {
    rmsnorm_kernel<<<SEQn, 256, 0, stream>>>(x, norm1_g + (size_t)i * HIDn,
                                             hbuf, HIDn);
    gemm_bf16<EPI_NONE, PERM_NONE><<<GRID(SEQn, QKVn), 256, 0, stream>>>(
        hbuf, qkv_w + (size_t)i * HIDn * QKVn, qkv_b + (size_t)i * QKVn,
        nullptr, qkvb, SEQn, QKVn, HIDn);
    rope_apply_kernel<<<SEQn, 256, 0, stream>>>(qkvb, cosT, sinT);
    int win = (i == 3 || i == 7) ? SEQn : 64;
    attn_kernel<<<dim3(SEQn / 64, HEADSn), 128, 0, stream>>>(qkvb, obuf, win);
    gemm_bf16<EPI_RES, PERM_NONE><<<GRID(SEQn, HIDn), 256, 0, stream>>>(
        obuf, proj_w + (size_t)i * HIDn * HIDn, proj_b + (size_t)i * HIDn, x,
        x, SEQn, HIDn, HIDn);
    rmsnorm_kernel<<<SEQn, 256, 0, stream>>>(x, norm2_g + (size_t)i * HIDn,
                                             hbuf, HIDn);
    gemm_bf16<EPI_SILU, PERM_NONE><<<GRID(SEQn, INTn), 256, 0, stream>>>(
        hbuf, gate_w + (size_t)i * HIDn * INTn, gate_b + (size_t)i * INTn,
        nullptr, gbuf, SEQn, INTn, HIDn);
    gemm_bf16<EPI_MUL, PERM_NONE><<<GRID(SEQn, INTn), 256, 0, stream>>>(
        hbuf, up_w + (size_t)i * HIDn * INTn, up_b + (size_t)i * INTn, gbuf,
        gbuf, SEQn, INTn, HIDn);
    gemm_bf16<EPI_RES, PERM_NONE><<<GRID(SEQn, HIDn), 256, 0, stream>>>(
        gbuf, down_w + (size_t)i * INTn * HIDn, down_b + (size_t)i * HIDn, x,
        x, SEQn, HIDn, INTn);
    if (i == 2 || i == 5 || i == 7) {  // deepstack (postshuffle=True)
      rmsnorm_kernel<<<MROWSn, 256, 0, stream>>>(
          x, ds_norm_g + (size_t)ds_slot * MHIDn, hbuf, MHIDn);
      gemm_bf16<EPI_GELU, PERM_NONE><<<GRID(MROWSn, MHIDn), 256, 0, stream>>>(
          hbuf, ds_l1_w + (size_t)ds_slot * MHIDn * MHIDn,
          ds_l1_b + (size_t)ds_slot * MHIDn, nullptr, obuf, MROWSn, MHIDn,
          MHIDn);
      gemm_bf16<EPI_NONE, PERM_MERGED>
          <<<GRID(MROWSn, OUTn), 256, 0, stream>>>(
              obuf, ds_l2_w + (size_t)ds_slot * MHIDn * OUTn,
              ds_l2_b + (size_t)ds_slot * OUTn, nullptr,
              out + (size_t)(1 + ds_slot) * MROWSn * OUTn, MROWSn, OUTn,
              MHIDn);
      ++ds_slot;
    }
  }

  // final merger (postshuffle=False): rms over 1152, contiguous view as 4608
  rmsnorm_kernel<<<SEQn, 256, 0, stream>>>(x, m_norm_g, hbuf, HIDn);
  gemm_bf16<EPI_GELU, PERM_NONE><<<GRID(MROWSn, MHIDn), 256, 0, stream>>>(
      hbuf, m_l1_w, m_l1_b, nullptr, obuf, MROWSn, MHIDn, MHIDn);
  gemm_bf16<EPI_NONE, PERM_MERGED><<<GRID(MROWSn, OUTn), 256, 0, stream>>>(
      obuf, m_l2_w, m_l2_b, nullptr, out, MROWSn, OUTn, MHIDn);
#undef GRID
}
